// ScaledDotProductAttention_52965536694578
// MI455X (gfx1250) — matrix operand order repaired and checked by reading, not compiled, not run on hardware
//
#include <hip/hip_runtime.h>
#include <hip/hip_bf16.h>

typedef _Float16 half16 __attribute__((ext_vector_type(16)));
typedef _Float16 half8  __attribute__((ext_vector_type(8)));
typedef float    float8 __attribute__((ext_vector_type(8)));

#define ATT_B 2
#define ATT_H 16
#define ATT_L 2048
#define ATT_D 64
#define NEG_MASK (-30000.0f)   // behaves like -1e9 through exp (underflows to 0); all-masked rows stay uniform

static __device__ __forceinline__ float8 wmma_f16(half16 a, half16 b, float8 c) {
    // v_wmma_f32_16x16x32_f16  (8-arg form: neg_a, A, neg_b, B, c_mod, C, reuse_a, reuse_b)
    return __builtin_amdgcn_wmma_f32_16x16x32_f16(false, a, false, b, (short)0, c, false, false);
}

__global__ __launch_bounds__(256)
void sdpa_kernel(const float* __restrict__ q, const float* __restrict__ k,
                 const float* __restrict__ v, const int* __restrict__ mask,
                 float* __restrict__ out, float* __restrict__ attn) {
    // One workgroup = one (b, h, 16-row query tile). 8 waves of 32.
    __shared__ _Float16 sP[16 * ATT_L];     // 64 KB: scores, then probs (f16)
    __shared__ float    sRed[4][32][8];     // 4 KB : cross-wave partial accumulators

    const int tid  = threadIdx.x;
    const int lane = tid & 31;
    // SGPR-ize the wave id so all wave-level control flow is scalar (no exec-mask loops).
    const int wave = __builtin_amdgcn_readfirstlane(tid >> 5);

    const int q0 = blockIdx.x * 16;
    const int h  = blockIdx.y;
    const int b  = blockIdx.z;

    const size_t bh = (size_t)b * ATT_H + h;
    const float* __restrict__ qbase = q + (bh * ATT_L + q0) * ATT_D;
    const float* __restrict__ kbase = k + bh * ATT_L * ATT_D;
    const float* __restrict__ vbase = v + bh * ATT_L * ATT_D;
    const int*   __restrict__ mbase = mask + (size_t)b * ATT_L * ATT_L + (size_t)q0 * ATT_L;
    float* __restrict__ attnbase = attn + (bh * ATT_L + q0) * ATT_L;
    float* __restrict__ outbase  = out  + (bh * ATT_L + q0) * ATT_D;

    const int hi   = lane >> 4;       // 0: lanes 0-15, 1: lanes 16-31
    const int mrow = lane & 15;       // M (A/D) or N (B/D) index
    const int aoff = hi * 8;          // A-matrix K offset / D-matrix M offset

    // ---- Load Q tile once as WMMA A operand (pre-scaled by 1/TEMPERATURE) ----
    // A layout (16-bit, 16x32): lane holds row M=lane&15; K(e) = aoff + e + (e>=8 ? 8 : 0)
    half16 a0, a1;
#pragma unroll
    for (int e = 0; e < 16; ++e) {
        const int kf = aoff + e + ((e >= 8) ? 8 : 0);
        a0[e] = (_Float16)(qbase[mrow * ATT_D + kf]      * 0.125f);
        a1[e] = (_Float16)(qbase[mrow * ATT_D + 32 + kf] * 0.125f);
    }

    // =========== Phase 1: S = (Q/T)·K^T, masked, -> LDS (f16) ===========
    // B layout (16-bit, 32x16): lane holds column N=lane&15; K(e) = 8*hi + e + (e>=8 ? 8 : 0)
    // Fixed trip count (16) so the compiler emits a scalar / unrollable loop.
    for (int i = 0; i < (ATT_L / 16) / 8; ++i) {
        const int kt  = (i << 3) + wave;
        const int key = kt * 16 + mrow;               // this lane's key row (B column)
        const float* __restrict__ krow = kbase + (size_t)key * ATT_D;
        const int bko = hi * 8;
        half16 b0, b1;
#pragma unroll
        for (int e = 0; e < 16; ++e) {
            b0[e] = (_Float16)krow[bko + e + ((e >= 8) ? 8 : 0)];          // features 0..31
            b1[e] = (_Float16)krow[32 + bko + e + ((e >= 8) ? 8 : 0)];     // features 32..63
        }
        float8 c = {0.f, 0.f, 0.f, 0.f, 0.f, 0.f, 0.f, 0.f};
        c = wmma_f16(a0, b0, c);
        c = wmma_f16(a1, b1, c);

        // D layout: VGPR r -> M = r + 8*hi, N = lane&15
        const int col = kt * 16 + mrow;
#pragma unroll
        for (int r = 0; r < 8; ++r) {
            const int m  = r + aoff;
            const int mv = mbase[(size_t)m * ATT_L + col];
            const float s = (mv == 0) ? NEG_MASK : c[r];
            sP[m * ATT_L + col] = (_Float16)s;
        }
    }
    __syncthreads();

    // =========== Phase 2: row softmax (16 lanes per row), write attn ===========
    {
        const int row = tid >> 4;         // 0..15  (wave w owns rows 2w, 2w+1)
        const int c16 = tid & 15;
        const _Float16* __restrict__ prow = &sP[row * ATT_L];

        float mx = -3.0e38f;
#pragma unroll 4
        for (int j = 0; j < ATT_L / 16; ++j)
            mx = fmaxf(mx, (float)prow[c16 + (j << 4)]);
#pragma unroll
        for (int sh = 8; sh >= 1; sh >>= 1)
            mx = fmaxf(mx, __shfl_xor(mx, sh, 16));

        float sum = 0.f;
#pragma unroll 4
        for (int j = 0; j < ATT_L / 16; ++j)
            sum += __expf((float)prow[c16 + (j << 4)] - mx);
#pragma unroll
        for (int sh = 8; sh >= 1; sh >>= 1)
            sum += __shfl_xor(sum, sh, 16);

        const float inv = 1.0f / sum;
        float* __restrict__ arow = attnbase + (size_t)row * ATT_L;
#pragma unroll 4
        for (int j = 0; j < ATT_L / 16; ++j) {
            const int col = c16 + (j << 4);
            const float p = __expf((float)prow[col] - mx) * inv;
            sP[row * ATT_L + col] = (_Float16)p;   // in-place: probs for phase 3
            arow[col] = p;                          // full attention output
        }
    }
    __syncthreads();

    // =========== Phase 3: out = P·V (P f16 from LDS, V f16 from global) ===========
    const int nt = wave & 3;   // feature tile: cols nt*16..nt*16+15   (SGPR)
    const int hv = wave >> 2;  // key half: 0 -> keys [0,1024), 1 -> [1024,2048)  (SGPR)

    float8 acc = {0.f, 0.f, 0.f, 0.f, 0.f, 0.f, 0.f, 0.f};
    const int f = nt * 16 + mrow;   // this lane's V feature column (B column)
    for (int i = 0; i < 32; ++i) {
        const int kb = (hv * 32 + i) * 32;   // 32-key chunk base

        // A operand: P rows from LDS, two 16B-aligned 8-half loads
        const _Float16* __restrict__ ap = &sP[mrow * ATT_L + kb + aoff];
        const half8 lo = *(const half8*)ap;
        const half8 hc = *(const half8*)(ap + 16);
        const half16 a = __builtin_shufflevector(lo, hc,
            0, 1, 2, 3, 4, 5, 6, 7, 8, 9, 10, 11, 12, 13, 14, 15);

        // B operand: V chunk, K(e) = 8*hi + e + (e>=8 ? 8 : 0)
        half16 bb;
        const int kstart = kb + hi * 8;
#pragma unroll
        for (int e = 0; e < 16; ++e)
            bb[e] = (_Float16)vbase[(size_t)(kstart + e + ((e >= 8) ? 8 : 0)) * ATT_D + f];

        acc = wmma_f16(a, bb, acc);
    }

    if (hv == 1) {
#pragma unroll
        for (int r = 0; r < 8; ++r) sRed[nt][lane][r] = acc[r];
    }
    __syncthreads();
    if (hv == 0) {
#pragma unroll
        for (int r = 0; r < 8; ++r) {
            const int m = r + aoff;
            outbase[(size_t)m * ATT_D + f] = acc[r] + sRed[nt][lane][r];
        }
    }
}

extern "C" void kernel_launch(void* const* d_in, const int* in_sizes, int n_in,
                              void* d_out, int out_size, void* d_ws, size_t ws_size,
                              hipStream_t stream) {
    (void)in_sizes; (void)n_in; (void)d_ws; (void)ws_size; (void)out_size;
    const float* q    = (const float*)d_in[0];
    const float* k    = (const float*)d_in[1];
    const float* v    = (const float*)d_in[2];
    const int*   mask = (const int*)d_in[3];

    float* out  = (float*)d_out;                                        // [B,H,L,D]
    float* attn = out + (size_t)ATT_B * ATT_H * ATT_L * ATT_D;          // [B,H,L,L]

    dim3 grid(ATT_L / 16, ATT_H, ATT_B);   // 128 x 16 x 2 = 4096 workgroups
    dim3 block(256);                        // 8 waves (wave32)
    sdpa_kernel<<<grid, block, 0, stream>>>(q, k, v, mask, out, attn);
}
